// Net_85272280694875
// MI455X (gfx1250) — compile-verified, matrix operand order checked by reading, not run on hardware
//
#include <hip/hip_runtime.h>
#include <math.h>

// ---------------------------------------------------------------------------
// GIN inference for MI455X (gfx1250, wave32, WMMA).
//   h1 = relu((x + Agg(x)) @ W1 + b1)        K=128 -> 64
//   h2 = relu((h1 + Agg(h1)) @ W2 + b2)      K=64  -> 64
//   h3 = relu(h2 @ Wf1 + bf1)                K=64  -> 64
//   out = log_softmax(h3 @ Wf2 + bf2)        K=64  -> 40 (padded to 64)
// Aggregation = scatter-add over edges (memory/atomic bound, ~2.5 GB traffic,
// hot set fits in 192MB L2). GEMMs run on v_wmma_f32_16x16x32_f16 (f32 accum).
// ---------------------------------------------------------------------------

typedef __attribute__((ext_vector_type(16))) _Float16 v16h;
typedef __attribute__((ext_vector_type(8)))  _Float16 v8h;
typedef __attribute__((ext_vector_type(8)))  float    v8f;

#define N_FIN   128
#define N_DIM   64
#define N_CLS   40

// ---- zero fill (float4 granularity) ---------------------------------------
__global__ void gin_zero_f4(float4* __restrict__ p, int n4) {
    int i = blockIdx.x * blockDim.x + threadIdx.x;
    if (i < n4) p[i] = make_float4(0.f, 0.f, 0.f, 0.f);
}

// ---- weight convert: W[k][Nsrc] f32 -> Wt[n][K] f16 (pad cols to Npad) -----
__global__ void gin_convert_w(const float* __restrict__ W, _Float16* __restrict__ Wt,
                              int K, int Nsrc, int Npad) {
    int i = blockIdx.x * blockDim.x + threadIdx.x;
    if (i >= K * Npad) return;
    int n = i / K;
    int k = i - n * K;
    Wt[n * K + k] = (n < Nsrc) ? (_Float16)W[k * Nsrc + n] : (_Float16)0.0f;
}

__global__ void gin_pad_bias(const float* __restrict__ b, float* __restrict__ bp,
                             int nsrc, int npad) {
    int i = threadIdx.x;
    if (i < npad) bp[i] = (i < nsrc) ? b[i] : 0.f;
}

// ---- edge scatter-add: AGG[dst] += H[src], F = 4 << f4shift floats/row -----
__global__ void gin_scatter(const float* __restrict__ H, const int* __restrict__ src,
                            const int* __restrict__ dst, float* __restrict__ AGG,
                            int nEdges, int f4shift) {
    unsigned tid = blockIdx.x * blockDim.x + threadIdx.x;
    unsigned e = tid >> f4shift;
    if (e >= (unsigned)nEdges) return;
    unsigned c = tid & ((1u << f4shift) - 1u);
    int F = 4 << f4shift;
    int s = src[e];
    int d = dst[e];
    const float4 v = ((const float4*)(H + (size_t)s * F))[c];
    float* o = AGG + (size_t)d * F + (c << 2);
    atomicAdd(o + 0, v.x);
    atomicAdd(o + 1, v.y);
    atomicAdd(o + 2, v.z);
    atomicAdd(o + 3, v.w);
}

// ---- fused GEMM: OUT = act((X [+ AGG]) @ Wt^T + bias) ----------------------
// One wave computes a 16-row x 64-col strip: 4 accumulators (N-tiles of 16),
// K swept in chunks of 32 via v_wmma_f32_16x16x32_f16.
// A-fragment lane layout (ISA 7.12.2, 16-bit A 16x32):
//   lanes 0-15 : row = lane,     K = {kc+0..7, kc+16..23}
//   lanes 16-31: row = lane-16,  K = {kc+8..15, kc+24..31}
// B symmetric with column index; Wt is pre-transposed [n][K] f16 so each
// half-fragment is one contiguous 16B load.
__launch_bounds__(256)
__global__ void gin_gemm(const float* __restrict__ X, const float* __restrict__ AGG,
                         const _Float16* __restrict__ Wt, const float* __restrict__ bias,
                         float* __restrict__ OUT, int M, int K,
                         int outStride, int nStore, int doRelu) {
    const int lane    = threadIdx.x & 31;
    const int wave    = threadIdx.x >> 5;
    const int rowBase = (blockIdx.x * 8 + wave) * 16;
    if (rowBase >= M) return;                 // wave-uniform: EXEC stays all-1s
    const int hi   = lane >> 4;               // 0 or 1
    const int l15  = lane & 15;
    const int khi  = hi * 8;
    const int mrow = rowBase + l15;

    v8f acc[4] = {};

    for (int kc = 0; kc < K; kc += 32) {
        // ---- A fragment: (X + AGG) row mrow, converted to f16 ----
        const float* xp = X + (size_t)mrow * K + kc + khi;
        float4 x0 = *(const float4*)(xp + 0);
        float4 x1 = *(const float4*)(xp + 4);
        float4 x2 = *(const float4*)(xp + 16);
        float4 x3 = *(const float4*)(xp + 20);
        if (AGG) {
            const float* gp = AGG + (size_t)mrow * K + kc + khi;
            float4 g0 = *(const float4*)(gp + 0);
            float4 g1 = *(const float4*)(gp + 4);
            float4 g2 = *(const float4*)(gp + 16);
            float4 g3 = *(const float4*)(gp + 20);
            x0.x += g0.x; x0.y += g0.y; x0.z += g0.z; x0.w += g0.w;
            x1.x += g1.x; x1.y += g1.y; x1.z += g1.z; x1.w += g1.w;
            x2.x += g2.x; x2.y += g2.y; x2.z += g2.z; x2.w += g2.w;
            x3.x += g3.x; x3.y += g3.y; x3.z += g3.z; x3.w += g3.w;
        }
        v16h a;
        a[0]  = (_Float16)x0.x; a[1]  = (_Float16)x0.y; a[2]  = (_Float16)x0.z; a[3]  = (_Float16)x0.w;
        a[4]  = (_Float16)x1.x; a[5]  = (_Float16)x1.y; a[6]  = (_Float16)x1.z; a[7]  = (_Float16)x1.w;
        a[8]  = (_Float16)x2.x; a[9]  = (_Float16)x2.y; a[10] = (_Float16)x2.z; a[11] = (_Float16)x2.w;
        a[12] = (_Float16)x3.x; a[13] = (_Float16)x3.y; a[14] = (_Float16)x3.z; a[15] = (_Float16)x3.w;

        // ---- 4 N-tiles reuse the A fragment ----
        #pragma unroll
        for (int t = 0; t < 4; ++t) {
            const _Float16* wp = Wt + (size_t)(16 * t + l15) * K + kc + khi;
            v8h b0 = *(const v8h*)(wp + 0);
            v8h b1 = *(const v8h*)(wp + 16);
            v16h b;
            #pragma unroll
            for (int j = 0; j < 8; ++j) { b[j] = b0[j]; b[8 + j] = b1[j]; }
            acc[t] = __builtin_amdgcn_wmma_f32_16x16x32_f16(
                false, a, false, b, (short)0, acc[t], false, false);
        }
    }

    // ---- bias + relu + store (C/D layout: vgpr r -> m = r + 8*hi) ----
    #pragma unroll
    for (int t = 0; t < 4; ++t) {
        int col = 16 * t + l15;
        if (col < nStore) {
            float bv = bias[col];
            #pragma unroll
            for (int r = 0; r < 8; ++r) {
                int m = rowBase + hi * 8 + r;
                float v = acc[t][r] + bv;
                if (doRelu) v = fmaxf(v, 0.f);
                OUT[(size_t)m * outStride + col] = v;
            }
        }
    }
}

// ---- in-place log_softmax over 40 classes, one wave32 per row --------------
__global__ void gin_logsoftmax(float* __restrict__ out, int M) {
    int lane = threadIdx.x & 31;
    int row  = (blockIdx.x * blockDim.x + threadIdx.x) >> 5;
    if (row >= M) return;
    float* p = out + (size_t)row * N_CLS;
    float v0 = p[lane];                                   // lanes 0..31 (<40)
    float v1 = (lane < 8) ? p[lane + 32] : -INFINITY;     // lanes 0..7 -> 32..39
    float m = fmaxf(v0, v1);
    #pragma unroll
    for (int off = 16; off >= 1; off >>= 1) m = fmaxf(m, __shfl_xor(m, off, 32));
    float s = expf(v0 - m) + ((lane < 8) ? expf(v1 - m) : 0.f);
    #pragma unroll
    for (int off = 16; off >= 1; off >>= 1) s += __shfl_xor(s, off, 32);
    float l = m + logf(s);
    p[lane] = v0 - l;
    if (lane < 8) p[lane + 32] = v1 - l;
}

// ---------------------------------------------------------------------------
extern "C" void kernel_launch(void* const* d_in, const int* in_sizes, int n_in,
                              void* d_out, int out_size, void* d_ws, size_t ws_size,
                              hipStream_t stream) {
    const float* x   = (const float*)d_in[0];
    const int*   ei  = (const int*)d_in[1];
    const float* W1  = (const float*)d_in[2];
    const float* b1  = (const float*)d_in[3];
    const float* W2  = (const float*)d_in[4];
    const float* b2  = (const float*)d_in[5];
    const float* Wf1 = (const float*)d_in[6];
    const float* bf1 = (const float*)d_in[7];
    const float* Wf2 = (const float*)d_in[8];
    const float* bf2 = (const float*)d_in[9];

    const int N = in_sizes[0] / N_FIN;       // 100000
    const int E = in_sizes[1] / 2;           // 1600000
    const int* src = ei;
    const int* dst = ei + E;

    // ---- workspace layout (agg[N*128] doubles as {agg2[N*64] | h2[N*64]}) ----
    float*     agg  = (float*)d_ws;                       // N*128 f32
    float*     h2   = agg + (size_t)N * N_DIM;            // second half of agg buf
    float*     hA   = agg + (size_t)N * N_FIN;            // N*64 f32 (h1, then h3)
    _Float16*  W1t  = (_Float16*)(hA + (size_t)N * N_DIM);// 64*128
    _Float16*  W2t  = W1t + 64 * 128;                     // 64*64
    _Float16*  Wf1t = W2t + 64 * 64;                      // 64*64
    _Float16*  Wf2t = Wf1t + 64 * 64;                     // 64*64 (cols 40..63 zero)
    float*     bf2p = (float*)(Wf2t + 64 * 64);           // 64 f32

    float* outp = (float*)d_out;

    // ---- one-time (per call) weight conversion: f32 row-major -> f16 [n][K] ----
    gin_convert_w<<<(64 * 128 + 255) / 256, 256, 0, stream>>>(W1,  W1t,  128, 64, 64);
    gin_convert_w<<<(64 * 64  + 255) / 256, 256, 0, stream>>>(W2,  W2t,  64,  64, 64);
    gin_convert_w<<<(64 * 64  + 255) / 256, 256, 0, stream>>>(Wf1, Wf1t, 64,  64, 64);
    gin_convert_w<<<(64 * 64  + 255) / 256, 256, 0, stream>>>(Wf2, Wf2t, 64,  N_CLS, 64);
    gin_pad_bias<<<1, 64, 0, stream>>>(bf2, bf2p, N_CLS, 64);

    const int gemmBlocks = ((N / 16) + 7) / 8;

    // ---- layer 1: agg128 + fused gemm (K=128) ----
    {
        int n4 = N * (N_FIN / 4);
        gin_zero_f4<<<(n4 + 255) / 256, 256, 0, stream>>>((float4*)agg, n4);
        long long tt = (long long)E << 5;   // f4shift = 5 (128 floats/row)
        gin_scatter<<<(unsigned)((tt + 255) / 256), 256, 0, stream>>>(x, src, dst, agg, E, 5);
        gin_gemm<<<gemmBlocks, 256, 0, stream>>>(x, agg, W1t, b1, hA,
                                                 N, N_FIN, N_DIM, N_DIM, 1);
    }

    // ---- layer 2: agg64 + fused gemm (K=64) ----
    {
        int n4 = N * (N_DIM / 4);
        gin_zero_f4<<<(n4 + 255) / 256, 256, 0, stream>>>((float4*)agg, n4);
        long long tt = (long long)E << 4;   // f4shift = 4 (64 floats/row)
        gin_scatter<<<(unsigned)((tt + 255) / 256), 256, 0, stream>>>(hA, src, dst, agg, E, 4);
        gin_gemm<<<gemmBlocks, 256, 0, stream>>>(hA, agg, W2t, b2, h2,
                                                 N, N_DIM, N_DIM, N_DIM, 1);
    }

    // ---- fc1 (relu), fc2 (logits into d_out), log_softmax in place ----
    gin_gemm<<<gemmBlocks, 256, 0, stream>>>(h2, nullptr, Wf1t, bf1, hA,
                                             N, N_DIM, N_DIM, N_DIM, 1);
    gin_gemm<<<gemmBlocks, 256, 0, stream>>>(hA, nullptr, Wf2t, bf2p, outp,
                                             N, N_DIM, N_CLS, N_CLS, 0);
    gin_logsoftmax<<<(N * 32 + 255) / 256, 256, 0, stream>>>(outp, N);
}
